// FourierGNN_65635690218057
// MI455X (gfx1250) — compile-verified, matrix operand order checked by reading, not run on hardware
//
#include <hip/hip_runtime.h>
#include <math.h>

// ---------------------------------------------------------------------------
// FourierGNN for MI455X (gfx1250). Seq-len-1 transformer => MHA collapses to
// two dense layers (Wv, out_w). All big GEMMs run on V_WMMA_F32_16X16X32_BF16
// with the A tile staged block-wide into LDS via GLOBAL_LOAD_ASYNC_TO_LDS.
// ---------------------------------------------------------------------------

#define N_NODES 10000
#define T_LEN   256
#define E_EDGES 160000
#define EA_EDGES (E_EDGES + N_NODES)   // + self loops
#define HID     128
#define HEADS   4
#define DFF     2048

typedef __attribute__((ext_vector_type(16))) __bf16 v16bf;
typedef __attribute__((ext_vector_type(8)))  __bf16 v8bf;
typedef __attribute__((ext_vector_type(8)))  float  v8f;

#ifndef __has_builtin
#define __has_builtin(x) 0
#endif

#if __has_builtin(__builtin_amdgcn_global_load_async_to_lds_b64)
#define USE_ASYNC_LDS 1
#else
#define USE_ASYNC_LDS 0
#endif

// b64 async payload type per clang signature: v2i with explicit address spaces
typedef int v2i __attribute__((vector_size(8)));
typedef __attribute__((address_space(1))) v2i glb_v2i;   // global source
typedef __attribute__((address_space(3))) v2i lds_v2i;   // LDS destination

__device__ __forceinline__ void wait_asynccnt0() {
#if __has_builtin(__builtin_amdgcn_s_wait_asynccnt)
    __builtin_amdgcn_s_wait_asynccnt(0);
#elif defined(__gfx1250__)
    asm volatile("s_wait_asynccnt 0" ::: "memory");
#endif
}

// ------------------------- order-preserving float<->uint --------------------
__device__ __forceinline__ unsigned fenc(float f) {
    unsigned u = __float_as_uint(f);
    return (u & 0x80000000u) ? ~u : (u | 0x80000000u);
}
__device__ __forceinline__ float fdec(unsigned u) {
    unsigned v = (u & 0x80000000u) ? (u & 0x7fffffffu) : ~u;
    return __uint_as_float(v);
}

// ------------------------------ DFT features --------------------------------
// feats_bf[n, 0:256]   = waveform (bf16)
// feats_bf[n, 256:512] = |FFT(waveform)| (bf16)
__global__ __launch_bounds__(256) void dft_feats_kernel(
    const float* __restrict__ w, __bf16* __restrict__ feats_bf)
{
    __shared__ float ws[256], ct[256], st[256];
    const int n = blockIdx.x, t = threadIdx.x;
    ws[t] = w[(size_t)n * 256 + t];
    const float th = (2.0f * 3.14159265358979323846f / 256.0f) * (float)t;
    ct[t] = cosf(th);
    st[t] = sinf(th);
    __syncthreads();
    float cr = 0.f, ci = 0.f;
    const int k = t;
    #pragma unroll 4
    for (int tt = 0; tt < 256; ++tt) {
        const int idx = (k * tt) & 255;
        const float wv = ws[tt];
        cr = fmaf(wv, ct[idx], cr);
        ci = fmaf(wv, st[idx], ci);
    }
    const float mag = sqrtf(cr * cr + ci * ci);
    const size_t base = (size_t)n * 512;
    feats_bf[base + k]       = (__bf16)ws[k];
    feats_bf[base + 256 + k] = (__bf16)mag;
}

// ------------------------------ small utils ---------------------------------
__global__ void fill_u32_kernel(unsigned* __restrict__ p, unsigned v, long long n)
{
    long long i = (long long)blockIdx.x * blockDim.x + threadIdx.x;
    if (i < n) p[i] = v;
}

__global__ void cast_bf16_kernel(const float* __restrict__ src,
                                 __bf16* __restrict__ dst, long long n)
{
    long long i = (long long)blockIdx.x * blockDim.x + threadIdx.x;
    if (i < n) dst[i] = (__bf16)src[i];
}

// src is (R x C) row major, dst is (C x R) row major, bf16
__global__ void transpose_cast_kernel(const float* __restrict__ src,
                                      __bf16* __restrict__ dst, int R, int C)
{
    long long i = (long long)blockIdx.x * blockDim.x + threadIdx.x;
    if (i >= (long long)R * C) return;
    const int r = (int)(i / C), c = (int)(i % C);
    dst[(size_t)c * R + r] = (__bf16)src[i];
}

// --------------------------- bf16 WMMA GEMM ---------------------------------
// C[M x Nw] = A[M x K] @ Bt[Nw x K]^T (+bias) (relu?) ; f32 out, optional bf16
// Grid: (M/16, Nw/128). Block: 256 threads = 8 waves, one 16x16 tile per wave.
// M % 16 == 0, Nw % 128 == 0, K % 64 == 0. No lane divergence => EXEC all 1s.
// Per 64-wide K step: the block stages the shared 16x64 A tile into LDS with
// one async-load instruction per wave (ASYNCcnt), all 8 waves read fragments
// back with ds_load_b128, while B fragments stream straight from global.
#define A_STRIDE 72   // 64 bf16 + 8 pad (144B rows, 16B aligned)

__global__ __launch_bounds__(256) void gemm_bf16_wmma_kernel(
    const __bf16* __restrict__ A, const __bf16* __restrict__ Bt,
    const float* __restrict__ bias, float* __restrict__ C,
    __bf16* __restrict__ Cbf, int M, int Nw, int K, int relu)
{
    __shared__ __align__(16) __bf16 As[16 * A_STRIDE];

    const int tid    = threadIdx.x;
    const int lane   = tid & 31;
    const int wave   = tid >> 5;
    const int lane16 = lane & 15;
    const int half   = lane >> 4;

    const int rowBase = blockIdx.x * 16;
    const int colBase = blockIdx.y * 128 + wave * 16;

    // cooperative A staging: thread t covers row=t>>4, 8-byte segment t&15
    const int srow = tid >> 4, sseg = tid & 15;
    const __bf16* agrow = A + (size_t)(rowBase + srow) * K + sseg * 4;
    __bf16* asl = &As[srow * A_STRIDE + sseg * 4];

    // CDNA5 16-bit A/B fragment layout: lane holds M(or N)=lane16; two
    // contiguous 8-elem K runs at half*8 and 16+half*8 within each 32-K step.
    const __bf16* afrag = &As[lane16 * A_STRIDE + half * 8];
    const __bf16* brow  = Bt + (size_t)(colBase + lane16) * K + half * 8;

    v8f acc = {};
    for (int k0 = 0; k0 < K; k0 += 64) {
#if USE_ASYNC_LDS
        __builtin_amdgcn_global_load_async_to_lds_b64(
            (glb_v2i*)(agrow + k0), (lds_v2i*)asl, 0, 0);
        wait_asynccnt0();
#else
        *(unsigned long long*)asl = *(const unsigned long long*)(agrow + k0);
#endif
        __syncthreads();

        if (k0 + 64 < K) {   // -> global_prefetch_b8
            __builtin_prefetch((const void*)(brow + k0 + 64), 0, 1);
            __builtin_prefetch((const void*)(agrow + k0 + 64), 0, 1);
        }

        #pragma unroll
        for (int s = 0; s < 2; ++s) {
            const v8bf a0 = *(const v8bf*)(afrag + s * 32);        // ds_load_b128
            const v8bf a1 = *(const v8bf*)(afrag + s * 32 + 16);
            const v8bf b0 = *(const v8bf*)(brow + k0 + s * 32);
            const v8bf b1 = *(const v8bf*)(brow + k0 + s * 32 + 16);
            const v16bf av = __builtin_shufflevector(a0, a1,
                0,1,2,3,4,5,6,7,8,9,10,11,12,13,14,15);
            const v16bf bv = __builtin_shufflevector(b0, b1,
                0,1,2,3,4,5,6,7,8,9,10,11,12,13,14,15);
            acc = __builtin_amdgcn_wmma_f32_16x16x32_bf16(
                      /*neg_a=*/false, av, /*neg_b=*/false, bv,
                      /*c_mod=*/(short)0, acc,
                      /*reuse_a=*/false, /*reuse_b=*/false);
        }
        __syncthreads();   // protect As from next iteration's staging
    }

    const int col = colBase + lane16;
    const float bval = bias ? bias[col] : 0.f;
    #pragma unroll
    for (int r = 0; r < 8; ++r) {
        const int row = rowBase + half * 8 + r;   // C/D layout: lanes=N, vgpr=M
        float v = acc[r] + bval;
        if (relu) v = fmaxf(v, 0.f);
        C[(size_t)row * Nw + col] = v;
        if (Cbf) Cbf[(size_t)row * Nw + col] = (__bf16)v;
    }
}

// ------------------------------- layernorm ----------------------------------
// out = LN(x + res?) * g + b over D=256 columns. One block per row.
__global__ __launch_bounds__(256) void layernorm_kernel(
    const float* __restrict__ x, const float* __restrict__ res,
    const float* __restrict__ g, const float* __restrict__ b,
    float* __restrict__ out, __bf16* __restrict__ outbf, int D)
{
    __shared__ float s1[256], s2[256];
    const int row = blockIdx.x, t = threadIdx.x;
    float y = x[(size_t)row * D + t];
    if (res) y += res[(size_t)row * D + t];
    s1[t] = y; s2[t] = y * y;
    __syncthreads();
    for (int s = 128; s > 0; s >>= 1) {
        if (t < s) { s1[t] += s1[t + s]; s2[t] += s2[t + s]; }
        __syncthreads();
    }
    const float mean = s1[0] / (float)D;
    const float var  = s2[0] / (float)D - mean * mean;
    const float inv  = rsqrtf(var + 1e-5f);
    const float o = (y - mean) * inv * g[t] + b[t];
    out[(size_t)row * D + t] = o;
    if (outbf) outbf[(size_t)row * D + t] = (__bf16)o;
}

// ------------------------------- GAT pieces ---------------------------------
__global__ void gat_dots_kernel(const float* __restrict__ H,
                                const float* __restrict__ att_s,
                                const float* __restrict__ att_d,
                                float* __restrict__ asrc, float* __restrict__ adst,
                                int n_nodes, int heads, int hd)
{
    const int gid = blockIdx.x * blockDim.x + threadIdx.x;
    if (gid >= n_nodes * heads) return;
    const int n = gid / heads, h = gid % heads;
    const float* hp = H + ((size_t)n * heads + h) * hd;
    const float* as = att_s + (size_t)h * hd;
    const float* ad = att_d + (size_t)h * hd;
    float s = 0.f, d = 0.f;
    for (int j = 0; j < hd; ++j) { const float v = hp[j]; s += v * as[j]; d += v * ad[j]; }
    asrc[gid] = s; adst[gid] = d;
}

__device__ __forceinline__ void edge_sd(const int* ei, int eid, int E, int& s, int& d)
{
    if (eid < E) { s = ei[eid]; d = ei[E + eid]; }
    else         { s = eid - E; d = eid - E; }          // self loops
}

__global__ void edge_max_kernel(const int* __restrict__ ei,
                                const float* __restrict__ asrc,
                                const float* __restrict__ adst,
                                unsigned* __restrict__ mx, int E, int EA, int heads)
{
    const long long gid = (long long)blockIdx.x * blockDim.x + threadIdx.x;
    if (gid >= (long long)EA * heads) return;
    const int eid = (int)(gid / heads), h = (int)(gid % heads);
    int s, d; edge_sd(ei, eid, E, s, d);
    float e = asrc[s * heads + h] + adst[d * heads + h];
    e = (e > 0.f) ? e : 0.2f * e;                       // leaky_relu
    atomicMax(&mx[d * heads + h], fenc(e));
}

__global__ void edge_expsum_kernel(const int* __restrict__ ei,
                                   const float* __restrict__ asrc,
                                   const float* __restrict__ adst,
                                   const unsigned* __restrict__ mx,
                                   float* __restrict__ den, float* __restrict__ eexp,
                                   int E, int EA, int heads)
{
    const long long gid = (long long)blockIdx.x * blockDim.x + threadIdx.x;
    if (gid >= (long long)EA * heads) return;
    const int eid = (int)(gid / heads), h = (int)(gid % heads);
    int s, d; edge_sd(ei, eid, E, s, d);
    float e = asrc[s * heads + h] + adst[d * heads + h];
    e = (e > 0.f) ? e : 0.2f * e;
    const float ee = __expf(e - fdec(mx[d * heads + h]));
    eexp[gid] = ee;
    atomicAdd(&den[d * heads + h], ee);
}

__global__ void edge_agg_kernel(const int* __restrict__ ei,
                                const float* __restrict__ H,
                                const float* __restrict__ eexp,
                                const float* __restrict__ den,
                                float* __restrict__ out,
                                int E, int EA, int heads, int hd)
{
    const long long gid = (long long)blockIdx.x * blockDim.x + threadIdx.x;
    if (gid >= (long long)EA * hd) return;
    const int eid = (int)(gid / hd), j = (int)(gid % hd);
    int s, d; edge_sd(ei, eid, E, s, d);
    for (int h = 0; h < heads; ++h) {
        const float alpha = eexp[(size_t)eid * heads + h] / den[d * heads + h];
        atomicAdd(&out[((size_t)d * heads + h) * hd + j],
                  H[((size_t)s * heads + h) * hd + j] * alpha);
    }
}

__global__ void bias_act_kernel(float* __restrict__ x, const float* __restrict__ bias,
                                __bf16* __restrict__ xbf, long long n, int D, int relu)
{
    const long long gid = (long long)blockIdx.x * blockDim.x + threadIdx.x;
    if (gid >= n) return;
    float v = x[gid] + bias[(int)(gid % D)];
    if (relu) v = fmaxf(v, 0.f);
    x[gid] = v;
    if (xbf) xbf[gid] = (__bf16)v;
}

// --------------------------- head: concat / fc2 -----------------------------
__global__ void concat_cast_kernel(const float* __restrict__ x2,
                                   const float* __restrict__ tf,
                                   __bf16* __restrict__ comb, int n_nodes)
{
    const long long gid = (long long)blockIdx.x * blockDim.x + threadIdx.x;
    if (gid >= (long long)n_nodes * 384) return;
    const int n = (int)(gid / 384), j = (int)(gid % 384);
    const float v = (j < 128) ? x2[(size_t)n * 128 + j] : tf[(size_t)n * 256 + (j - 128)];
    comb[gid] = (__bf16)v;
}

__global__ void fc2_kernel(const float* __restrict__ h1, const float* __restrict__ w2,
                           const float* __restrict__ b2, float* __restrict__ out, int n_nodes)
{
    const int n = blockIdx.x * blockDim.x + threadIdx.x;
    if (n >= n_nodes) return;
    float s = b2[0];
    for (int i = 0; i < 128; ++i) s += h1[(size_t)n * 128 + i] * w2[i];
    out[n] = s;
}

// ---------------------------------------------------------------------------
extern "C" void kernel_launch(void* const* d_in, const int* in_sizes, int n_in,
                              void* d_out, int out_size, void* d_ws, size_t ws_size,
                              hipStream_t stream)
{
    (void)in_sizes; (void)n_in; (void)out_size; (void)ws_size;
    // Input order (recursive insertion order of setup_inputs()):
    // 0 waveforms, 1 edge_index,
    // 2-5  gat1: W, att_src, att_dst, bias
    // 6-9  gat2: W, att_src, att_dst, bias
    // enc layer l (l=0,1) base 10+12l: sa.in_w,in_b,out_w,out_b, ln1.g,b, ln2.g,b,
    //                                  lin1_w, lin1_b, lin2_w, lin2_b
    // 34,35 enc_norm g,b
    // dec layer l (l=0,1) base 36+18l: sa(4), ca(4), ln1(2), ln2(2), ln3(2),
    //                                  lin1_w, lin1_b, lin2_w, lin2_b
    // 72,73 dec_norm g,b ; 74 fc1_w, 75 fc1_b, 76 fc2_w, 77 fc2_b
    auto F = [&](int i) { return (const float*)d_in[i]; };
    const float* wave = F(0);
    const int*   ei   = (const int*)d_in[1];
    float* outp = (float*)d_out;

    const int N = N_NODES, E = E_EDGES, EA = EA_EDGES;

    // ---------------- deterministic bump allocator over d_ws ----------------
    char* W = (char*)d_ws;
    size_t off = 0;
    auto alloc = [&](size_t bytes) -> void* {
        off = (off + 255) & ~(size_t)255;
        void* p = (void*)(W + off);
        off += bytes;
        return p;
    };
    float*  mid32 = (float*) alloc((size_t)N * 2048 * 4);  // FFN mid (f32)
    float*  h32   = (float*) alloc((size_t)N * 256 * 4);   // enc hidden  | GAT1 H lower
    float*  t32   = (float*) alloc((size_t)N * 256 * 4);   // dec hidden  | GAT1 H upper
    float*  tmp32 = (float*) alloc((size_t)N * 256 * 4);   // scratch     | GAT1 x lower
    float*  tmp2  = (float*) alloc((size_t)N * 256 * 4);   // scratch     | GAT1 x upper
    float*  mem32 = (float*) alloc((size_t)N * 256 * 4);   // enc memory  | GAT2 H
    __bf16* midbf = (__bf16*)alloc((size_t)N * 2048 * 2);  // FFN mid bf  | feats_bf / x1bf
    __bf16* hbf   = (__bf16*)alloc((size_t)N * 256 * 2);
    __bf16* tbf   = (__bf16*)alloc((size_t)N * 256 * 2);
    __bf16* tmpbf = (__bf16*)alloc((size_t)N * 256 * 2);
    __bf16* membf = (__bf16*)alloc((size_t)N * 256 * 2);
    __bf16* wslab = (__bf16*)alloc((size_t)2048 * 256 * 2);// weight staging (reused)
    float*  x2    = (float*) alloc((size_t)N * 128 * 4);   // GAT2 output
    float*  h1f   = (float*) alloc((size_t)N * 128 * 4);   // fc1 output
    __bf16* combbf= (__bf16*)alloc((size_t)N * 384 * 2);
    float*  asrc1 = (float*) alloc((size_t)N * HEADS * 4);
    float*  adst1 = (float*) alloc((size_t)N * HEADS * 4);
    unsigned* mx1 = (unsigned*)alloc((size_t)N * HEADS * 4);
    float*  den1  = (float*) alloc((size_t)N * HEADS * 4);
    float*  eexp1 = (float*) alloc((size_t)EA * HEADS * 4);
    float*  asrc2 = (float*) alloc((size_t)N * 4);
    float*  adst2 = (float*) alloc((size_t)N * 4);
    unsigned* mx2 = (unsigned*)alloc((size_t)N * 4);
    float*  den2  = (float*) alloc((size_t)N * 4);
    float*  eexp2 = (float*) alloc((size_t)EA * 4);
    // Lifetime aliases (GAT phase finishes before the transformer phase):
    __bf16* featsbf = midbf;   // (N x 512) bf16
    float*  H1      = h32;     // (N x 512) f32 spans h32..t32 (contiguous)
    float*  x1      = tmp32;   // (N x 512) f32 spans tmp32..tmp2 (contiguous)
    __bf16* x1bf    = midbf;   // reused after featsbf is dead
    float*  H2      = mem32;   // (N x 128) f32

    auto cdiv = [](long long a, long long b) { return (unsigned)((a + b - 1) / b); };
    auto fill = [&](void* p, unsigned v, long long n) {
        fill_u32_kernel<<<cdiv(n, 256), 256, 0, stream>>>((unsigned*)p, v, n);
    };
    auto cast = [&](const float* s, __bf16* d, long long n) {
        cast_bf16_kernel<<<cdiv(n, 256), 256, 0, stream>>>(s, d, n);
    };
    auto gemm = [&](const __bf16* A, const __bf16* Bt, const float* bias,
                    float* C, __bf16* Cbf, int M, int Nw, int K, int relu) {
        dim3 g((unsigned)(M / 16), (unsigned)(Nw / 128));
        gemm_bf16_wmma_kernel<<<g, 256, 0, stream>>>(A, Bt, bias, C, Cbf, M, Nw, K, relu);
    };
    auto ln = [&](const float* x, const float* res, const float* g, const float* b,
                  float* o, __bf16* obf) {
        layernorm_kernel<<<N, 256, 0, stream>>>(x, res, g, b, o, obf, 256);
    };

    // ===================== Stage 1: DFT features (bf16) =====================
    dft_feats_kernel<<<N, 256, 0, stream>>>(wave, featsbf);

    // ===================== Stage 2: GAT layer 1 (4 heads) ===================
    transpose_cast_kernel<<<cdiv(512 * 512, 256), 256, 0, stream>>>(F(2), wslab, 512, 512);
    gemm(featsbf, wslab, nullptr, H1, nullptr, N, 512, 512, 0);
    gat_dots_kernel<<<cdiv((long long)N * HEADS, 256), 256, 0, stream>>>(
        H1, F(3), F(4), asrc1, adst1, N, HEADS, HID);
    fill(mx1, 0u, (long long)N * HEADS);
    fill(den1, 0u, (long long)N * HEADS);
    fill(x1, 0u, (long long)N * 512);
    edge_max_kernel<<<cdiv((long long)EA * HEADS, 256), 256, 0, stream>>>(
        ei, asrc1, adst1, mx1, E, EA, HEADS);
    edge_expsum_kernel<<<cdiv((long long)EA * HEADS, 256), 256, 0, stream>>>(
        ei, asrc1, adst1, mx1, den1, eexp1, E, EA, HEADS);
    edge_agg_kernel<<<cdiv((long long)EA * HID, 256), 256, 0, stream>>>(
        ei, H1, eexp1, den1, x1, E, EA, HEADS, HID);
    bias_act_kernel<<<cdiv((long long)N * 512, 256), 256, 0, stream>>>(
        x1, F(5), x1bf, (long long)N * 512, 512, 1);   // + relu (writes x1bf)

    // ===================== Stage 3: GAT layer 2 (1 head) ====================
    transpose_cast_kernel<<<cdiv(512 * 128, 256), 256, 0, stream>>>(F(6), wslab, 512, 128);
    gemm(x1bf, wslab, nullptr, H2, nullptr, N, 128, 512, 0);
    gat_dots_kernel<<<cdiv((long long)N, 256), 256, 0, stream>>>(
        H2, F(7), F(8), asrc2, adst2, N, 1, HID);
    fill(mx2, 0u, N);
    fill(den2, 0u, N);
    fill(x2, 0u, (long long)N * 128);
    edge_max_kernel<<<cdiv((long long)EA, 256), 256, 0, stream>>>(
        ei, asrc2, adst2, mx2, E, EA, 1);
    edge_expsum_kernel<<<cdiv((long long)EA, 256), 256, 0, stream>>>(
        ei, asrc2, adst2, mx2, den2, eexp2, E, EA, 1);
    edge_agg_kernel<<<cdiv((long long)EA * HID, 256), 256, 0, stream>>>(
        ei, H2, eexp2, den2, x2, E, EA, 1, HID);
    bias_act_kernel<<<cdiv((long long)N * 128, 256), 256, 0, stream>>>(
        x2, F(9), nullptr, (long long)N * 128, 128, 0);  // no relu after gat2

    // ================= Stage 4: transformer (seq len == 1) ==================
    const int dd = 256 * 256;   // 256x256 weight elems; Wv slice = in_w + 2*dd
    cast(wave, hbf, (long long)N * 256);
    const float* hprev = wave;
    for (int l = 0; l < 2; ++l) {             // encoder
        const int b = 10 + 12 * l;
        // self-attn (collapsed): v = h@Wv.T+bv ; o = v@out_w.T+out_b
        cast(F(b + 0) + 2 * dd, wslab, dd);
        gemm(hbf, wslab, F(b + 1) + 512, tmp32, tmpbf, N, 256, 256, 0);
        cast(F(b + 2), wslab, dd);
        gemm(tmpbf, wslab, F(b + 3), tmp2, nullptr, N, 256, 256, 0);
        ln(tmp2, hprev, F(b + 4), F(b + 5), h32, hbf);
        hprev = h32;
        // FFN
        cast(F(b + 8), wslab, (long long)DFF * 256);
        gemm(hbf, wslab, F(b + 9), mid32, midbf, N, DFF, 256, 1);
        cast(F(b + 10), wslab, (long long)256 * DFF);
        gemm(midbf, wslab, F(b + 11), tmp2, nullptr, N, 256, DFF, 0);
        ln(tmp2, h32, F(b + 6), F(b + 7), h32, hbf);
    }
    ln(h32, nullptr, F(34), F(35), mem32, membf);       // encoder memory

    cast(wave, tbf, (long long)N * 256);
    const float* tprev = wave;
    for (int l = 0; l < 2; ++l) {             // decoder
        const int b = 36 + 18 * l;
        // self-attn
        cast(F(b + 0) + 2 * dd, wslab, dd);
        gemm(tbf, wslab, F(b + 1) + 512, tmp32, tmpbf, N, 256, 256, 0);
        cast(F(b + 2), wslab, dd);
        gemm(tmpbf, wslab, F(b + 3), tmp2, nullptr, N, 256, 256, 0);
        ln(tmp2, tprev, F(b + 8), F(b + 9), t32, tbf);
        tprev = t32;
        // cross-attn (kv = mem)
        cast(F(b + 4) + 2 * dd, wslab, dd);
        gemm(membf, wslab, F(b + 5) + 512, tmp32, tmpbf, N, 256, 256, 0);
        cast(F(b + 6), wslab, dd);
        gemm(tmpbf, wslab, F(b + 7), tmp2, nullptr, N, 256, 256, 0);
        ln(tmp2, t32, F(b + 10), F(b + 11), t32, tbf);
        // FFN
        cast(F(b + 14), wslab, (long long)DFF * 256);
        gemm(tbf, wslab, F(b + 15), mid32, midbf, N, DFF, 256, 1);
        cast(F(b + 16), wslab, (long long)256 * DFF);
        gemm(midbf, wslab, F(b + 17), tmp2, nullptr, N, 256, DFF, 0);
        ln(tmp2, t32, F(b + 12), F(b + 13), t32, tbf);
    }
    ln(t32, nullptr, F(72), F(73), tmp32, nullptr);     // tfeat -> tmp32

    // ======================= Stage 5: head ==================================
    concat_cast_kernel<<<cdiv((long long)N * 384, 256), 256, 0, stream>>>(
        x2, tmp32, combbf, N);
    cast(F(74), wslab, (long long)128 * 384);
    gemm(combbf, wslab, F(75), h1f, nullptr, N, 128, 384, 1);
    fc2_kernel<<<cdiv((long long)N, 256), 256, 0, stream>>>(h1f, F(76), F(77), outp, N);
}
